// RoIHead_41944650612769
// MI455X (gfx1250) — matrix operand order repaired and checked by reading, not compiled, not run on hardware
//
#include <hip/hip_runtime.h>

// ---------------------------------------------------------------------------
// RoI head for Fast R-CNN on MI455X (gfx1250, wave32).
// Roofline: streaming W1 (411 MB fp32) at 23.3 TB/s (~18us) dominates; bf16
// WMMA (v_wmma_f32_16x16x32_bf16, f32 accum) keeps matrix compute far above
// the HBM roof, so the GEMMs pipeline global->LDS traffic:
//   - A tiles: CDNA5 async DMA (global_load_async_to_lds_b128 / ASYNCcnt)
//   - B tiles: nontemporal fp32 loads -> in-register bf16 convert -> LDS
// Double-buffered LDS, one barrier per K-step. Deterministic split-K.
// ---------------------------------------------------------------------------

#define FC    512
#define FH    37
#define FW    50
#define RNUM  128
#define KPOOL 25088   // 512*7*7
#define DH    4096
#define LDSP  40      // LDS row pitch in bf16 elems (80B, 16B-aligned, padded)

typedef __attribute__((ext_vector_type(16))) __bf16       v16bf;
typedef __attribute__((ext_vector_type(8)))  float        v8f;
typedef __attribute__((ext_vector_type(8)))  unsigned int v8u;
typedef __attribute__((ext_vector_type(4)))  float        fx4;   // native vec4

static __device__ __forceinline__ unsigned short f2bf(float f) {
  unsigned int u = __float_as_uint(f);
  u = (u + 0x7FFFu + ((u >> 16) & 1u)) >> 16;   // RNE fp32 -> bf16
  return (unsigned short)u;
}

static __device__ __forceinline__ v16bf frag16(uint4 a, uint4 b) {
  v8u t;
  t[0] = a.x; t[1] = a.y; t[2] = a.z; t[3] = a.w;
  t[4] = b.x; t[5] = b.y; t[6] = b.z; t[7] = b.w;
  return __builtin_bit_cast(v16bf, t);
}

// LDS byte offset of a __shared__ object: flat shared addresses carry the LDS
// offset in addr[31:0] (ISA 10.2 aperture mapping), which is what the async
// load-to-LDS VDST operand wants.
static __device__ __forceinline__ unsigned ldsaddr(const void* p) {
  return (unsigned)(size_t)p;
}

static __device__ __forceinline__ void async_a_b128(unsigned lds_dst,
                                                    unsigned byte_off,
                                                    const unsigned short* base) {
  asm volatile("global_load_async_to_lds_b128 %0, %1, %2"
               :: "v"(lds_dst), "v"(byte_off), "s"(base)
               : "memory");
}

// ---------------------------------------------------------------------------
// 1) RoI pool: x[1,512,37,50] + rois[128,4] -> bf16 A[128, 25088]
// ---------------------------------------------------------------------------
__global__ void roipool_kernel(const float* __restrict__ x,
                               const float* __restrict__ rois,
                               unsigned short* __restrict__ poolA) {
  const int r    = blockIdx.y;
  const int kidx = blockIdx.x * 256 + threadIdx.x;   // 98*256 == 25088 exactly
  const float y1 = rois[r*4+0], x1 = rois[r*4+1];
  const float y2 = rois[r*4+2], x2 = rois[r*4+3];
  const int h0 = (int)(y1 * 0.0625f);
  const int h1 = (int)(y2 * 0.0625f) + 1;
  const int w0 = (int)(x1 * 0.0625f);
  const int w1 = (int)(x2 * 0.0625f) + 1;
  const int Lh = h1 - h0, Lw = w1 - w0;

  const int c   = kidx / 49;
  const int rem = kidx - c * 49;
  const int ph  = rem / 7;
  const int pw  = rem - ph * 7;

  int hs  = h0 + (ph * Lh) / 7;
  int he  = h0 + ((ph + 1) * Lh + 6) / 7;
  int wsx = w0 + (pw * Lw) / 7;
  int we  = w0 + ((pw + 1) * Lw + 6) / 7;
  if (he > FH) he = FH;
  if (we > FW) we = FW;

  float m = -3.402823466e+38f;
  const float* xp = x + (size_t)c * (FH * FW);
  for (int yy = hs; yy < he; ++yy)
    for (int xx = wsx; xx < we; ++xx)
      m = fmaxf(m, xp[yy * FW + xx]);

  poolA[(size_t)r * KPOOL + kidx] = f2bf(m);
}

// ---------------------------------------------------------------------------
// 2) Main WMMA GEMM: part[ks][128][N] = A_bf16[128,Kchunk] @ B_f32[K,N]
//    256 thr (8 wave32), tile M=128 x N=64, K step 32, double-buffered LDS,
//    async A DMA + register-pipelined B, one barrier per step.
// ---------------------------------------------------------------------------
__global__ void __launch_bounds__(256)
gemm_bf16_wmma(const unsigned short* __restrict__ A,
               const float* __restrict__ B,
               float* __restrict__ part,
               int K, int N, int KCHUNK) {
  __shared__ __align__(16) unsigned short Alds[2][128 * LDSP];  // 2 x 10 KB
  __shared__ __align__(16) unsigned short Blds[2][64 * LDSP];   // 2 x 5 KB

  const int tid  = threadIdx.x;
  const int w    = tid >> 5;          // wave id (wave32)
  const int lane = tid & 31;
  const int lrow = lane & 15;
  const int hi   = lane >> 4;
  const int ks     = blockIdx.y;
  const int k0     = ks * KCHUNK;
  const int nbase  = blockIdx.x * 64;
  const int nsteps = KCHUNK >> 5;

  // A staging: this thread DMAs rows am and am+64, bf16 sub-chunk ap..ap+7
  const int am = tid >> 2;
  const int ap = (tid & 3) << 3;
  const unsigned aoff[2][2] = {
      { ldsaddr(&Alds[0][am * LDSP + ap]), ldsaddr(&Alds[0][(am + 64) * LDSP + ap]) },
      { ldsaddr(&Alds[1][am * LDSP + ap]), ldsaddr(&Alds[1][(am + 64) * LDSP + ap]) } };
  const unsigned abyte0 = (unsigned)(((size_t)am * K + ap) * 2);
  const unsigned abyte1 = (unsigned)((((size_t)am + 64) * K + ap) * 2);

  // B staging: this thread loads rows bk and bk+16, n-quad bn..bn+3
  const int bk = tid >> 4;
  const int bn = (tid & 15) << 2;

  v8f acc[4] = {};

  // ---- prologue: tile 0 ----
  {
    const unsigned kb = (unsigned)k0 * 2;
    async_a_b128(aoff[0][0], abyte0 + kb, A);
    async_a_b128(aoff[0][1], abyte1 + kb, A);
  }
  fx4 br0 = __builtin_nontemporal_load(
      (const fx4*)(B + (size_t)(k0 + bk) * N + nbase + bn));
  fx4 br1 = __builtin_nontemporal_load(
      (const fx4*)(B + (size_t)(k0 + bk + 16) * N + nbase + bn));

  for (int s = 0; s < nsteps; ++s) {
    const int cur = s & 1;
    const int nxt = cur ^ 1;
    const int sn  = (s + 1 < nsteps) ? s + 1 : s;   // clamp: branch-free tail
    const int ktn = k0 + (sn << 5);

    // ---- pre-barrier: convert & store current B tile (transposed [n][k]) --
    {
      unsigned short* bl = Blds[cur];
      bl[(bn + 0) * LDSP + bk]      = f2bf(br0[0]);
      bl[(bn + 1) * LDSP + bk]      = f2bf(br0[1]);
      bl[(bn + 2) * LDSP + bk]      = f2bf(br0[2]);
      bl[(bn + 3) * LDSP + bk]      = f2bf(br0[3]);
      bl[(bn + 0) * LDSP + bk + 16] = f2bf(br1[0]);
      bl[(bn + 1) * LDSP + bk + 16] = f2bf(br1[1]);
      bl[(bn + 2) * LDSP + bk + 16] = f2bf(br1[2]);
      bl[(bn + 3) * LDSP + bk + 16] = f2bf(br1[3]);
    }
    asm volatile("s_wait_asynccnt 0x0" ::: "memory");  // tile s A is in LDS
    __syncthreads();

    // ---- post-barrier: issue next-tile A DMA + B register prefetch --------
    {
      const unsigned kb = (unsigned)ktn * 2;
      async_a_b128(aoff[nxt][0], abyte0 + kb, A);
      async_a_b128(aoff[nxt][1], abyte1 + kb, A);
    }
    const fx4 nr0 = __builtin_nontemporal_load(
        (const fx4*)(B + (size_t)(ktn + bk) * N + nbase + bn));
    const fx4 nr1 = __builtin_nontemporal_load(
        (const fx4*)(B + (size_t)(ktn + bk + 16) * N + nbase + bn));

    // ---- compute on tile s ------------------------------------------------
    const unsigned short* al  = Alds[cur];
    const unsigned short* blp = Blds[cur];
    const int abase = (16 * w + lrow) * LDSP + (hi << 3);
    const uint4 a0 = *(const uint4*)(al + abase);
    const uint4 a1 = *(const uint4*)(al + abase + 16);
    uint4 b0[4], b1[4];
#pragma unroll
    for (int nt = 0; nt < 4; ++nt) {
      const int bbase = (nt * 16 + lrow) * LDSP + (hi << 3);
      b0[nt] = *(const uint4*)(blp + bbase);
      b1[nt] = *(const uint4*)(blp + bbase + 16);
    }
    const v16bf af = frag16(a0, a1);
#pragma unroll
    for (int nt = 0; nt < 4; ++nt)
      acc[nt] = __builtin_amdgcn_wmma_f32_16x16x32_bf16(
          false, af, false, frag16(b0[nt], b1[nt]), (short)0, acc[nt],
          false, false);

    br0 = nr0; br1 = nr1;
  }

  // D layout: lane<16 -> M=r, lane>=16 -> M=8+r; N = lane&15
#pragma unroll
  for (int nt = 0; nt < 4; ++nt)
#pragma unroll
    for (int r = 0; r < 8; ++r) {
      const int m = 16 * w + (hi << 3) + r;
      const int n = nbase + nt * 16 + lrow;
      part[((size_t)ks * 128 + m) * N + n] = acc[nt][r];
    }
}

// ---------------------------------------------------------------------------
// 3) Reduce K-splits + bias + ReLU -> bf16 activations
// ---------------------------------------------------------------------------
__global__ void reduce_relu_bf16(const float* __restrict__ part,
                                 const float* __restrict__ bias,
                                 unsigned short* __restrict__ out,
                                 int N, int KS) {
  const int idx = blockIdx.x * 256 + threadIdx.x;   // exact grid
  const int m = idx / N, n = idx - m * N;
  float s = bias[n];
#pragma unroll 4
  for (int ks = 0; ks < KS; ++ks)
    s += part[((size_t)ks * 128 + m) * N + n];
  s = fmaxf(s, 0.f);
  out[idx] = f2bf(s);
}

// ---------------------------------------------------------------------------
// 4) Heads: fc7[128,4096] @ {Wloc[4096,84] | Wscore[4096,21]}
//    N-tiles of 16, zero-padded OOB columns; async A DMA, single buffer.
// ---------------------------------------------------------------------------
__global__ void __launch_bounds__(256)
head_gemm(const unsigned short* __restrict__ A,
          const float* __restrict__ Wloc,
          const float* __restrict__ Wscore,
          float* __restrict__ part, int KCHUNK) {
  __shared__ __align__(16) unsigned short Alds[128 * LDSP];
  __shared__ __align__(16) unsigned short Blds[16 * LDSP];

  const int tid  = threadIdx.x;
  const int w    = tid >> 5;
  const int lane = tid & 31;
  const int lrow = lane & 15;
  const int hi   = lane >> 4;
  const int bx = blockIdx.x;
  const int ks = blockIdx.y;
  const int k0 = ks * KCHUNK;
  const int K  = DH;

  const float* B; int Ncols, cb;
  if (bx < 6) { B = Wloc;   Ncols = 84; cb = bx * 16; }
  else        { B = Wscore; Ncols = 21; cb = (bx - 6) * 16; }

  const int am = tid >> 2;
  const int ap = (tid & 3) << 3;
  const unsigned aoff0 = ldsaddr(&Alds[am * LDSP + ap]);
  const unsigned aoff1 = ldsaddr(&Alds[(am + 64) * LDSP + ap]);
  const unsigned abyte0 = (unsigned)(((size_t)am * K + ap) * 2);
  const unsigned abyte1 = (unsigned)((((size_t)am + 64) * K + ap) * 2);

  const int bkk = tid >> 4;   // 0..15 (and +16)
  const int bnn = tid & 15;

  v8f acc = {};
  for (int kk = 0; kk < KCHUNK; kk += 32) {
    const int kt = k0 + kk;
    const unsigned kb = (unsigned)kt * 2;
    async_a_b128(aoff0, abyte0 + kb, A);
    async_a_b128(aoff1, abyte1 + kb, A);

    const int col = cb + bnn;
    const float v0 = (col < Ncols) ? B[(size_t)(kt + bkk) * Ncols + col] : 0.f;
    const float v1 = (col < Ncols) ? B[(size_t)(kt + bkk + 16) * Ncols + col] : 0.f;
    Blds[bnn * LDSP + bkk]      = f2bf(v0);
    Blds[bnn * LDSP + bkk + 16] = f2bf(v1);

    asm volatile("s_wait_asynccnt 0x0" ::: "memory");
    __syncthreads();

    const int abase = (16 * w + lrow) * LDSP + (hi << 3);
    const uint4 a0 = *(const uint4*)(&Alds[abase]);
    const uint4 a1 = *(const uint4*)(&Alds[abase + 16]);
    const int bbase = lrow * LDSP + (hi << 3);
    const uint4 b0 = *(const uint4*)(&Blds[bbase]);
    const uint4 b1 = *(const uint4*)(&Blds[bbase + 16]);
    acc = __builtin_amdgcn_wmma_f32_16x16x32_bf16(
        false, frag16(a0, a1), false, frag16(b0, b1), (short)0, acc,
        false, false);
    __syncthreads();
  }

#pragma unroll
  for (int r = 0; r < 8; ++r) {
    const int m = 16 * w + (hi << 3) + r;
    part[((size_t)ks * 128 + m) * 128 + bx * 16 + lrow] = acc[r];
  }
}

// ---------------------------------------------------------------------------
// 5) Head reduce -> d_out = [roi_locs(128*84) | roi_scores(128*21)] fp32
// ---------------------------------------------------------------------------
__global__ void head_reduce(const float* __restrict__ part,
                            const float* __restrict__ bloc,
                            const float* __restrict__ bscore,
                            float* __restrict__ out) {
  const int idx = blockIdx.x * 256 + threadIdx.x;
  const int NLOC = 128 * 84;
  if (idx >= NLOC + 128 * 21) return;
  int m, col; float b;
  if (idx < NLOC) {
    m = idx / 84; const int j = idx - m * 84; col = j;     b = bloc[j];
  } else {
    const int t = idx - NLOC;
    m = t / 21;  const int j = t - m * 21;  col = 96 + j;  b = bscore[j];
  }
  float s = b;
#pragma unroll
  for (int ks = 0; ks < 4; ++ks)
    s += part[((size_t)ks * 128 + m) * 128 + col];
  out[idx] = s;
}

// ---------------------------------------------------------------------------
extern "C" void kernel_launch(void* const* d_in, const int* in_sizes, int n_in,
                              void* d_out, int out_size, void* d_ws, size_t ws_size,
                              hipStream_t stream) {
  const float* x      = (const float*)d_in[0];
  const float* rois   = (const float*)d_in[1];
  const float* W1     = (const float*)d_in[2];
  const float* b1     = (const float*)d_in[3];
  const float* W2     = (const float*)d_in[4];
  const float* b2     = (const float*)d_in[5];
  const float* Wloc   = (const float*)d_in[6];
  const float* bloc   = (const float*)d_in[7];
  const float* Wscore = (const float*)d_in[8];
  const float* bscore = (const float*)d_in[9];
  float* out = (float*)d_out;

  char* ws = (char*)d_ws;
  size_t o = 0;
  auto alloc = [&](size_t bytes) -> void* {
    void* p = ws + o;
    o = (o + bytes + 255) & ~(size_t)255;
    return p;
  };
  unsigned short* poolA = (unsigned short*)alloc((size_t)RNUM * KPOOL * 2); // 6.4 MB
  float*          part6 = (float*)         alloc((size_t)4 * 128 * DH * 4); // 8.4 MB
  unsigned short* fc6   = (unsigned short*)alloc((size_t)128 * DH * 2);     // 1 MB
  float*          part7 = (float*)         alloc((size_t)4 * 128 * DH * 4); // 8.4 MB
  unsigned short* fc7   = (unsigned short*)alloc((size_t)128 * DH * 2);     // 1 MB
  float*          partH = (float*)         alloc((size_t)4 * 128 * 128 * 4);

  roipool_kernel<<<dim3(98, 128), 256, 0, stream>>>(x, rois, poolA);
  // fc6: K=25088, KS=4 (chunk 6272 = 196 K-steps), 64 N-tiles
  gemm_bf16_wmma<<<dim3(64, 4), 256, 0, stream>>>(poolA, W1, part6, KPOOL, DH, 6272);
  reduce_relu_bf16<<<dim3((128 * DH) / 256), 256, 0, stream>>>(part6, b1, fc6, DH, 4);
  // fc7: K=4096, KS=4 (chunk 1024 = 32 K-steps)
  gemm_bf16_wmma<<<dim3(64, 4), 256, 0, stream>>>(fc6, W2, part7, DH, DH, 1024);
  reduce_relu_bf16<<<dim3((128 * DH) / 256), 256, 0, stream>>>(part7, b2, fc7, DH, 4);
  // heads: (6 loc tiles + 2 score tiles) x KS=4
  head_gemm<<<dim3(8, 4), 256, 0, stream>>>(fc7, Wloc, Wscore, partH, 1024);
  head_reduce<<<dim3(53), 256, 0, stream>>>(partH, bloc, bscore, out);
}
